// GraphBeliefPropagationNN_2894807957505
// MI455X (gfx1250) — matrix operand order repaired and checked by reading, not compiled, not run on hardware
//
#include <hip/hip_runtime.h>

// ---------------------------------------------------------------------------
// Types for CDNA5 WMMA (wave32)
// ---------------------------------------------------------------------------
typedef __bf16 bf16_t;
typedef __attribute__((ext_vector_type(16))) __bf16 v16bf;
typedef __attribute__((ext_vector_type(8)))  __bf16 v8bf;
typedef __attribute__((ext_vector_type(8)))  float  v8f;

#define N_NODES 100000
#define N_EDGES 3200000
#define HIDDEN  256
#define IN_DIM  128
#define BP_ITERS 5

#define MBLK       32           // node rows per workgroup
#define LDS_STRIDE 264          // 256 + 8 bf16 pad: 528B row = 132 dwords -> bank-conflict-free

// f32 -> bf16: native fptrunc (round-to-nearest-even); lets the backend use
// the hardware bf16 convert ops instead of a 3-VALU integer emulation.
static __device__ __forceinline__ bf16_t f2bf(float f) {
  return (bf16_t)f;
}

static __device__ __forceinline__ v8f wmma_bf16(v16bf a, v16bf b, v8f c) {
  return __builtin_amdgcn_wmma_f32_16x16x32_bf16(false, a, false, b, (short)0, c,
                                                 false, false);
}

// ---------------------------------------------------------------------------
// Prep: convert + transpose weight  W[K x N] (f32, row major)  ->  Wt[N x K] bf16
// ---------------------------------------------------------------------------
__global__ void transpose_convert_kernel(const float* __restrict__ W,
                                         bf16_t* __restrict__ Wt, int K, int N) {
  int idx = blockIdx.x * blockDim.x + threadIdx.x;
  if (idx >= K * N) return;
  int n = idx / K;
  int k = idx - n * K;
  Wt[(size_t)n * K + k] = f2bf(W[(size_t)k * N + n]);
}

// Extract src column (edges[:,0]) into a compact array
__global__ void extract_src_kernel(const int* __restrict__ edges,
                                   int* __restrict__ src) {
  int e = blockIdx.x * blockDim.x + threadIdx.x;
  if (e < N_EDGES) src[e] = edges[2 * e];
}

// ---------------------------------------------------------------------------
// A-fragment loaders (16-bit A 16x32, contiguous K order matching B):
//   lanes 0-15 : K = {0..15};  lanes 16-31 : K = {16..31}
// ---------------------------------------------------------------------------
static __device__ __forceinline__ v16bf load_a_f32(const float* __restrict__ row,
                                                   int k0, int halfsel) {
  const float* p0 = row + k0 + halfsel * 16;
  const float* p1 = row + k0 + 8 + halfsel * 16;
  v16bf a;
#pragma unroll
  for (int i = 0; i < 8; ++i) { a[i] = f2bf(p0[i]); a[i + 8] = f2bf(p1[i]); }
  return a;
}

static __device__ __forceinline__ v16bf load_a_lds(const bf16_t* row, int k0,
                                                   int halfsel) {
  v8bf lo = *(const v8bf*)(row + k0 + halfsel * 16);
  v8bf hi = *(const v8bf*)(row + k0 + 8 + halfsel * 16);
  v16bf a;
#pragma unroll
  for (int i = 0; i < 8; ++i) { a[i] = lo[i]; a[i + 8] = hi[i]; }
  return a;
}

// ---------------------------------------------------------------------------
// Fused MLP kernel: X -> (3 WMMA GEMM layers, activations in LDS) -> head
//   grid: N_NODES/MBLK blocks of 256 threads (8 waves).
//   wave w: tileM = w>>2 (two 16-row sub-tiles), strip = w&3 (64-wide N strip)
//   => per layer each wave holds 4 accumulator tiles; A fragment reused 4x.
// ---------------------------------------------------------------------------
__global__ __launch_bounds__(256)
void fused_mlp_kernel(const float* __restrict__ X,
                      const bf16_t* __restrict__ Wt_in,   // [256 x 128] bf16
                      const float* __restrict__ b_in,
                      const bf16_t* __restrict__ Wt_h1,   // [256 x 256] bf16
                      const float* __restrict__ b_h1,
                      const bf16_t* __restrict__ Wt_h2,   // [256 x 256] bf16
                      const float* __restrict__ b_h2,
                      const float* __restrict__ W_out,    // [256 x 2] f32
                      const float* __restrict__ b_out,    // [2]
                      float* __restrict__ zlp)            // [N_NODES x 2]
{
  __shared__ bf16_t bufA[MBLK * LDS_STRIDE];   // h1 then h3
  __shared__ bf16_t bufB[MBLK * LDS_STRIDE];   // h2

  const int wave     = threadIdx.x >> 5;        // 0..7
  const int lane     = threadIdx.x & 31;
  const int halfsel  = lane >> 4;
  const int l15      = lane & 15;
  const int tileM    = wave >> 2;               // 0..1
  const int strip    = wave & 3;                // 0..3
  const int rowLocal = tileM * 16 + l15;        // local A row [0,32)
  const int rowGlob  = blockIdx.x * MBLK + rowLocal;
  const int stripN   = strip * 64;              // first N column of this strip

  v8f acc[4];

  // ---------------- layer 1:  h1 = X @ W_in + b_in   (K = 128) --------------
  {
#pragma unroll
    for (int t = 0; t < 4; ++t) acc[t] = (v8f){};
    const float* arow = X + (size_t)rowGlob * IN_DIM;
#pragma unroll
    for (int k0 = 0; k0 < IN_DIM; k0 += 32) {
      v16bf a = load_a_f32(arow, k0, halfsel);
      const bf16_t* bb = Wt_in + k0 + halfsel * 16;
#pragma unroll
      for (int t = 0; t < 4; ++t) {
        v16bf b = *(const v16bf*)(bb + (size_t)(stripN + t * 16 + l15) * IN_DIM);
        acc[t] = wmma_bf16(a, b, acc[t]);
      }
    }
#pragma unroll
    for (int t = 0; t < 4; ++t) {
      int nc = stripN + t * 16 + l15;
      float bv = b_in[nc];
#pragma unroll
      for (int r = 0; r < 8; ++r)
        bufA[(tileM * 16 + r + halfsel * 8) * LDS_STRIDE + nc] = f2bf(acc[t][r] + bv);
    }
  }
  __syncthreads();

  // ---------------- layer 2:  h2 = h1 @ W_h1 + b_h1  (K = 256) --------------
  {
#pragma unroll
    for (int t = 0; t < 4; ++t) acc[t] = (v8f){};
    const bf16_t* arow = &bufA[rowLocal * LDS_STRIDE];
#pragma unroll
    for (int k0 = 0; k0 < HIDDEN; k0 += 32) {
      v16bf a = load_a_lds(arow, k0, halfsel);
      const bf16_t* bb = Wt_h1 + k0 + halfsel * 16;
#pragma unroll
      for (int t = 0; t < 4; ++t) {
        v16bf b = *(const v16bf*)(bb + (size_t)(stripN + t * 16 + l15) * HIDDEN);
        acc[t] = wmma_bf16(a, b, acc[t]);
      }
    }
#pragma unroll
    for (int t = 0; t < 4; ++t) {
      int nc = stripN + t * 16 + l15;
      float bv = b_h1[nc];
#pragma unroll
      for (int r = 0; r < 8; ++r)
        bufB[(tileM * 16 + r + halfsel * 8) * LDS_STRIDE + nc] = f2bf(acc[t][r] + bv);
    }
  }
  __syncthreads();

  // ---------------- layer 3:  h3 = h2 @ W_h2 + b_h2  (K = 256) --------------
  {
#pragma unroll
    for (int t = 0; t < 4; ++t) acc[t] = (v8f){};
    const bf16_t* arow = &bufB[rowLocal * LDS_STRIDE];
#pragma unroll
    for (int k0 = 0; k0 < HIDDEN; k0 += 32) {
      v16bf a = load_a_lds(arow, k0, halfsel);
      const bf16_t* bb = Wt_h2 + k0 + halfsel * 16;
#pragma unroll
      for (int t = 0; t < 4; ++t) {
        v16bf b = *(const v16bf*)(bb + (size_t)(stripN + t * 16 + l15) * HIDDEN);
        acc[t] = wmma_bf16(a, b, acc[t]);
      }
    }
#pragma unroll
    for (int t = 0; t < 4; ++t) {
      int nc = stripN + t * 16 + l15;
      float bv = b_h2[nc];
#pragma unroll
      for (int r = 0; r < 8; ++r)
        bufA[(tileM * 16 + r + halfsel * 8) * LDS_STRIDE + nc] = f2bf(acc[t][r] + bv);
    }
  }
  __syncthreads();

  // ------------- head: relu(h3 @ W_out + b_out) -> log_softmax --------------
  if (threadIdx.x < MBLK) {
    const bf16_t* hp = &bufA[threadIdx.x * LDS_STRIDE];
    float a0 = b_out[0], a1 = b_out[1];
#pragma unroll 8
    for (int k = 0; k < HIDDEN; ++k) {
      float h = (float)hp[k];
      a0 = fmaf(h, W_out[2 * k + 0], a0);
      a1 = fmaf(h, W_out[2 * k + 1], a1);
    }
    a0 = fmaxf(a0, 0.f);
    a1 = fmaxf(a1, 0.f);
    float m = fmaxf(a0, a1);
    float l = m + __logf(__expf(a0 - m) + __expf(a1 - m));
    int n = blockIdx.x * MBLK + threadIdx.x;
    zlp[2 * n + 0] = a0 - l;
    zlp[2 * n + 1] = a1 - l;
  }
}

// ---------------------------------------------------------------------------
// BP kernels
// ---------------------------------------------------------------------------
__global__ void copy_f32_kernel(const float* __restrict__ src,
                                float* __restrict__ dst, int n) {
  int i = blockIdx.x * blockDim.x + threadIdx.x;
  if (i < n) dst[i] = src[i];
}

__global__ void edge_kernel(const int* __restrict__ src,     // [E] compact
                            const float* __restrict__ lp_prev,
                            float* __restrict__ lm,          // [E x 2] in/out
                            float* __restrict__ lp_next,     // pre-init to zlp
                            const float* __restrict__ H,     // [2 x 2]
                            int first) {
  int e = blockIdx.x * blockDim.x + threadIdx.x;
  if (e >= N_EDGES) return;
  int s = src[e];
  float t0 = lp_prev[2 * s + 0];
  float t1 = lp_prev[2 * s + 1];
  if (!first) {
    t0 -= lm[2 * e + 0];
    t1 -= lm[2 * e + 1];
  }
  float h00 = H[0], h01 = H[1], h10 = H[2], h11 = H[3];
  float a = h00 + t0, b = h01 + t1;
  float m0 = fmaxf(a, b);
  float nm0 = m0 + __logf(__expf(a - m0) + __expf(b - m0));
  a = h10 + t0; b = h11 + t1;
  float m1 = fmaxf(a, b);
  float nm1 = m1 + __logf(__expf(a - m1) + __expf(b - m1));
  lm[2 * e + 0] = nm0;
  lm[2 * e + 1] = nm1;
  atomicAdd(&lp_next[2 * s + 0], nm0);
  atomicAdd(&lp_next[2 * s + 1], nm1);
}

__global__ void norm_kernel(const float* __restrict__ lp, float* __restrict__ out) {
  int n = blockIdx.x * blockDim.x + threadIdx.x;
  if (n >= N_NODES) return;
  float a = lp[2 * n + 0], b = lp[2 * n + 1];
  float m = fmaxf(a, b);
  float l = m + __logf(__expf(a - m) + __expf(b - m));
  out[2 * n + 0] = a - l;
  out[2 * n + 1] = b - l;
}

// ---------------------------------------------------------------------------
// Orchestration
// ---------------------------------------------------------------------------
extern "C" void kernel_launch(void* const* d_in, const int* in_sizes, int n_in,
                              void* d_out, int out_size, void* d_ws, size_t ws_size,
                              hipStream_t stream) {
  (void)in_sizes; (void)n_in; (void)out_size; (void)ws_size;
  const float* X     = (const float*)d_in[0];
  const int*   edges = (const int*)d_in[1];
  const float* W_in  = (const float*)d_in[2];
  const float* b_in  = (const float*)d_in[3];
  const float* W_h1  = (const float*)d_in[4];
  const float* b_h1  = (const float*)d_in[5];
  const float* W_h2  = (const float*)d_in[6];
  const float* b_h2  = (const float*)d_in[7];
  const float* W_out = (const float*)d_in[8];
  const float* b_out = (const float*)d_in[9];
  const float* H     = (const float*)d_in[10];
  float* out = (float*)d_out;

  // ---- workspace carve-up (256B aligned) ----
  char* ws = (char*)d_ws;
  size_t off = 0;
  auto alloc = [&](size_t bytes) -> char* {
    char* p = ws + off;
    off += (bytes + 255) & ~size_t(255);
    return p;
  };
  bf16_t* Wt_in  = (bf16_t*)alloc((size_t)HIDDEN * IN_DIM * 2);   // 64 KB
  bf16_t* Wt_h1  = (bf16_t*)alloc((size_t)HIDDEN * HIDDEN * 2);   // 128 KB
  bf16_t* Wt_h2  = (bf16_t*)alloc((size_t)HIDDEN * HIDDEN * 2);   // 128 KB
  float*  zlp    = (float*)alloc((size_t)N_NODES * 2 * 4);        // 0.8 MB
  float*  lpA    = (float*)alloc((size_t)N_NODES * 2 * 4);
  float*  lpB    = (float*)alloc((size_t)N_NODES * 2 * 4);
  float*  lm     = (float*)alloc((size_t)N_EDGES * 2 * 4);        // 25.6 MB
  int*    srcIdx = (int*)alloc((size_t)N_EDGES * 4);              // 12.8 MB

  // ---- prep: transpose/convert weights; extract src column ----
  {
    int n1 = IN_DIM * HIDDEN;
    transpose_convert_kernel<<<(n1 + 255) / 256, 256, 0, stream>>>(W_in, Wt_in, IN_DIM, HIDDEN);
    int n2 = HIDDEN * HIDDEN;
    transpose_convert_kernel<<<(n2 + 255) / 256, 256, 0, stream>>>(W_h1, Wt_h1, HIDDEN, HIDDEN);
    transpose_convert_kernel<<<(n2 + 255) / 256, 256, 0, stream>>>(W_h2, Wt_h2, HIDDEN, HIDDEN);
    extract_src_kernel<<<(N_EDGES + 255) / 256, 256, 0, stream>>>(edges, srcIdx);
  }

  // ---- fused MLP + head (one pass, activations never touch HBM) ----
  fused_mlp_kernel<<<N_NODES / MBLK, 256, 0, stream>>>(
      X, Wt_in, b_in, Wt_h1, b_h1, Wt_h2, b_h2, W_out, b_out, zlp);

  // ---- belief propagation iterations ----
  const float* lp_prev = zlp;
  float* bufs[2] = {lpA, lpB};
  for (int it = 0; it < BP_ITERS; ++it) {
    float* lp_next = bufs[it & 1];
    copy_f32_kernel<<<(2 * N_NODES + 255) / 256, 256, 0, stream>>>(zlp, lp_next, 2 * N_NODES);
    edge_kernel<<<(N_EDGES + 255) / 256, 256, 0, stream>>>(srcIdx, lp_prev, lm, lp_next, H, it == 0);
    lp_prev = lp_next;
  }

  // ---- final normalization ----
  norm_kernel<<<(N_NODES + 255) / 256, 256, 0, stream>>>(lp_prev, out);
}
